// Block_46883863003250
// MI455X (gfx1250) — matrix operand order repaired and checked by reading, not compiled, not run on hardware
//
#include <hip/hip_runtime.h>
#include <hip/hip_fp16.h>

typedef __attribute__((ext_vector_type(16))) _Float16 v16h;
typedef __attribute__((ext_vector_type(8)))  float    v8f;

#define WMMA(a,b,c) __builtin_amdgcn_wmma_f32_16x16x32_f16(false,(a),false,(b),(short)0,(c),false,false)

// ---------------- fragment loaders (contiguous k order: k = 16*(lane/16) + element) ----------------
// A 16x32 (row-major source): lanes0-15 M=lane, K=0..15; lanes16-31 M=lane-16, K=16..31
static __device__ inline v16h load_a(const _Float16* p, int lda, int k0, int lane) {
    int kb = (lane < 16) ? 0 : 16;
    const _Float16* r = p + (size_t)(lane & 15) * lda + k0 + kb;
    v16h a;
#pragma unroll
    for (int i = 0; i < 8; i++) a[i] = r[i];
#pragma unroll
    for (int i = 0; i < 8; i++) a[8 + i] = r[8 + i];
    return a;
}
// B 32x16 from K-contiguous (N-major) storage: row n holds K values contiguously.
// lanes0-15 cover K=k0..k0+15, lanes16-31 cover K=k0+16..k0+31; N = n0 + (lane&15)
static __device__ inline v16h load_bk(const _Float16* p, int ld, int n0, int k0, int lane) {
    int kb = (lane < 16) ? 0 : 16;
    const _Float16* r = p + (size_t)(n0 + (lane & 15)) * ld + k0 + kb;
    v16h b;
#pragma unroll
    for (int i = 0; i < 16; i++) b[i] = r[i];
    return b;
}
// A frag from 16x16 hidden tile in LDS, zero-padded to K=32
static __device__ inline v16h load_hid_a(const _Float16* Hs, int lane) {
    const _Float16* r = Hs + (lane & 15) * 16;
    v16h a;
#pragma unroll
    for (int i = 0; i < 16; i++) a[i] = (lane < 16) ? r[i] : (_Float16)0.f;
    return a;
}
// B frag from w2^T [384][16] expert slab: each lane<16 reads its full 16-element row; K>=16 is zero.
static __device__ inline v16h load_b_w2t(const _Float16* p, int n0, int lane) {
    v16h b;
    if (lane < 16) {
        const _Float16* r = p + (size_t)(n0 + lane) * 16;
#pragma unroll
        for (int i = 0; i < 16; i++) b[i] = r[i];
    } else {
#pragma unroll
        for (int i = 0; i < 16; i++) b[i] = (_Float16)0.f;
    }
    return b;
}

static __device__ inline float red_max16(float v) {
#pragma unroll
    for (int m = 1; m < 16; m <<= 1) v = fmaxf(v, __shfl_xor(v, m, 32));
    return v;
}
static __device__ inline float red_sum16(float v) {
#pragma unroll
    for (int m = 1; m < 16; m <<= 1) v += __shfl_xor(v, m, 32);
    return v;
}
static __device__ inline float red_sum32(float v) {
#pragma unroll
    for (int m = 1; m < 32; m <<= 1) v += __shfl_xor(v, m, 32);
    return v;
}

// ---------------- kernels ----------------
// f32 [batch][R][C] -> f16 transposed [batch][C][R]
__global__ void cvtT_kernel(const float* __restrict__ s, _Float16* __restrict__ d,
                            int R, int Ccols, int n) {
    int i = blockIdx.x * blockDim.x + threadIdx.x;
    if (i >= n) return;
    int rc = R * Ccols;
    int b = i / rc, rem = i - b * rc;
    int r = rem / Ccols, c = rem - r * Ccols;
    d[(size_t)b * rc + (size_t)c * R + r] = (_Float16)s[i];
}

__global__ __launch_bounds__(256) void ln1_kernel(const float* __restrict__ x,
                                                  const float* __restrict__ g,
                                                  const float* __restrict__ b,
                                                  _Float16* __restrict__ h16) {
    int wave = threadIdx.x >> 5, lane = threadIdx.x & 31;
    int row = blockIdx.x * 8 + wave;
    const float* xr = x + (size_t)row * 384;
    float v[12], s = 0.f;
#pragma unroll
    for (int i = 0; i < 12; i++) { v[i] = xr[i * 32 + lane]; s += v[i]; }
    s = red_sum32(s);
    float mu = s * (1.f / 384.f), q = 0.f;
#pragma unroll
    for (int i = 0; i < 12; i++) { float d0 = v[i] - mu; q += d0 * d0; }
    q = red_sum32(q);
    float rstd = rsqrtf(q * (1.f / 384.f) + 1e-5f);
    _Float16* hr = h16 + (size_t)row * 384;
#pragma unroll
    for (int i = 0; i < 12; i++) {
        int c = i * 32 + lane;
        hr[c] = (_Float16)((v[i] - mu) * rstd * g[c] + b[c]);
    }
}

// fused Q/K/V: one wave per (b, head, m-tile, n-tile); A fragment reused for 3 WMMAs/k-step
__global__ __launch_bounds__(256) void qkv_kernel(const _Float16* __restrict__ h16,
                                                  const _Float16* __restrict__ wqT,
                                                  const _Float16* __restrict__ wkT,
                                                  const _Float16* __restrict__ wvT,
                                                  _Float16* __restrict__ q,
                                                  _Float16* __restrict__ k,
                                                  _Float16* __restrict__ v) {
    int gw = blockIdx.x * 8 + (threadIdx.x >> 5);
    int lane = threadIdx.x & 31;
    int nt = gw & 3;  gw >>= 2;
    int mt = gw & 15; gw >>= 4;
    int hh = gw % 6;  gw /= 6;
    int bb = gw;
    const _Float16* A = h16 + (size_t)(bb * 256 + mt * 16) * 384;
    size_t wo = (size_t)hh * 64 * 384;     // [H][D][C] transposed
    const _Float16* WQ = wqT + wo;
    const _Float16* WK = wkT + wo;
    const _Float16* WV = wvT + wo;
    v8f aq = {}, ak = {}, av = {};
    for (int k0 = 0; k0 < 384; k0 += 32) {
        if (k0 + 64 < 384)
            __builtin_prefetch(A + (size_t)(lane & 15) * 384 + k0 + 64, 0, 1);
        v16h a = load_a(A, 384, k0, lane);
        aq = WMMA(a, load_bk(WQ, 384, nt * 16, k0, lane), aq);
        ak = WMMA(a, load_bk(WK, 384, nt * 16, k0, lane), ak);
        av = WMMA(a, load_bk(WV, 384, nt * 16, k0, lane), av);
    }
    size_t base = ((size_t)(bb * 6 + hh) * 256 + mt * 16) * 64;
    int rowoff = (lane >= 16) ? 8 : 0, n = nt * 16 + (lane & 15);
#pragma unroll
    for (int r = 0; r < 8; r++) {
        size_t o = base + (size_t)(r + rowoff) * 64 + n;
        q[o] = (_Float16)aq[r];
        k[o] = (_Float16)ak[r];
        v[o] = (_Float16)av[r];
    }
}

__global__ __launch_bounds__(256) void attn_kernel(const _Float16* __restrict__ q,
                                                   const _Float16* __restrict__ kk,
                                                   const _Float16* __restrict__ vv,
                                                   _Float16* __restrict__ attn) {
    __shared__ _Float16 Ks[256 * 64];   // row-major [s][d]
    __shared__ _Float16 Vt[64 * 256];   // transposed [d][s]
    __shared__ _Float16 Ps[8][16 * 32];
    int bh = blockIdx.x;
    const _Float16* qb = q + (size_t)bh * 16384;
    const _Float16* kb = kk + (size_t)bh * 16384;
    const _Float16* vb = vv + (size_t)bh * 16384;
    // K tile: CDNA5 async global->LDS copy (ASYNCcnt path, no VGPR round-trip).
    // V tile: transposed during staging (sync path).
    for (int i = threadIdx.x; i < 2048; i += 256) {
        unsigned ldsK = (unsigned)(size_t)(const void*)(Ks + (size_t)i * 8);
        const _Float16* gK = kb + (size_t)i * 8;
        asm volatile("global_load_async_to_lds_b128 %0, %1, off"
                     :: "v"(ldsK), "v"(gK)
                     : "memory");
        uint4 raw = ((const uint4*)vb)[i];
        const _Float16* t = (const _Float16*)&raw;
        int s = i >> 3, d0 = (i & 7) * 8;
#pragma unroll
        for (int j = 0; j < 8; j++) Vt[(d0 + j) * 256 + s] = t[j];
    }
    asm volatile("s_wait_asynccnt 0x0" ::: "memory");
    __syncthreads();
    int wave = threadIdx.x >> 5, lane = threadIdx.x & 31;
    int rowoff = (lane >= 16) ? 8 : 0;
    int bidx = bh / 6, hh = bh % 6;
    for (int mt = wave; mt < 16; mt += 8) {
        int m0 = mt * 16;
        const _Float16* QA = qb + (size_t)m0 * 64;
        v16h aq0 = load_a(QA, 64, 0, lane);
        v16h aq1 = load_a(QA, 64, 32, lane);
        v8f o0 = {}, o1 = {}, o2 = {}, o3 = {};
        float mrow[8], lrow[8];
#pragma unroll
        for (int r = 0; r < 8; r++) { mrow[r] = -1e30f; lrow[r] = 0.f; }
        int npair = m0 / 32 + 1;
        for (int pr = 0; pr < npair; pr++) {
            int n0 = pr * 32;
            v8f s0 = {}, s1 = {};
            s0 = WMMA(aq0, load_bk(Ks, 64, n0, 0, lane), s0);
            s0 = WMMA(aq1, load_bk(Ks, 64, n0, 32, lane), s0);
            s1 = WMMA(aq0, load_bk(Ks, 64, n0 + 16, 0, lane), s1);
            s1 = WMMA(aq1, load_bk(Ks, 64, n0 + 16, 32, lane), s1);
            bool last = (pr == npair - 1);
            _Float16* pw = Ps[wave];
#pragma unroll
            for (int r = 0; r < 8; r++) {
                float x0 = s0[r] * 0.125f, x1 = s1[r] * 0.125f;
                if (last) {
                    int row = m0 + r + rowoff;
                    int c0 = n0 + (lane & 15);
                    if (c0 > row)      x0 = -1e30f;
                    if (c0 + 16 > row) x1 = -1e30f;
                }
                float tm = red_max16(fmaxf(x0, x1));
                float nm = fmaxf(mrow[r], tm);
                float alpha = __expf(mrow[r] - nm);
                float p0 = __expf(x0 - nm), p1 = __expf(x1 - nm);
                float rs = red_sum16(p0 + p1);
                lrow[r] = lrow[r] * alpha + rs;
                mrow[r] = nm;
                o0[r] *= alpha; o1[r] *= alpha; o2[r] *= alpha; o3[r] *= alpha;
                int prow = r + rowoff;
                pw[prow * 32 + (lane & 15)]      = (_Float16)p0;
                pw[prow * 32 + 16 + (lane & 15)] = (_Float16)p1;
            }
            v16h ap = load_a(Ps[wave], 32, 0, lane);
            o0 = WMMA(ap, load_bk(Vt, 256, 0,  n0, lane), o0);
            o1 = WMMA(ap, load_bk(Vt, 256, 16, n0, lane), o1);
            o2 = WMMA(ap, load_bk(Vt, 256, 32, n0, lane), o2);
            o3 = WMMA(ap, load_bk(Vt, 256, 48, n0, lane), o3);
        }
        _Float16* ob = attn + ((size_t)(bidx * 256 + m0) * 384) + hh * 64;
#pragma unroll
        for (int r = 0; r < 8; r++) {
            float inv = 1.f / lrow[r];
            size_t base = (size_t)(r + rowoff) * 384 + (lane & 15);
            ob[base +  0] = (_Float16)(o0[r] * inv);
            ob[base + 16] = (_Float16)(o1[r] * inv);
            ob[base + 32] = (_Float16)(o2[r] * inv);
            ob[base + 48] = (_Float16)(o3[r] * inv);
        }
    }
}

__global__ __launch_bounds__(256) void proj_kernel(const _Float16* __restrict__ attn,
                                                   const _Float16* __restrict__ wprojT,
                                                   const float* __restrict__ bproj,
                                                   const float* __restrict__ x,
                                                   float* __restrict__ x2) {
    int gw = blockIdx.x * 8 + (threadIdx.x >> 5);
    int lane = threadIdx.x & 31;
    int nt = gw % 24, mt = gw / 24;
    const _Float16* A = attn + (size_t)mt * 16 * 384;
    v8f acc = {};
    for (int k0 = 0; k0 < 384; k0 += 32) {
        if (k0 + 64 < 384)
            __builtin_prefetch(A + (size_t)(lane & 15) * 384 + k0 + 64, 0, 1);
        acc = WMMA(load_a(A, 384, k0, lane), load_bk(wprojT, 384, nt * 16, k0, lane), acc);
    }
    int rowoff = (lane >= 16) ? 8 : 0, n = nt * 16 + (lane & 15);
#pragma unroll
    for (int r = 0; r < 8; r++) {
        size_t idx = (size_t)(mt * 16 + r + rowoff) * 384 + n;
        x2[idx] = acc[r] + bproj[n] + x[idx];
    }
}

__global__ __launch_bounds__(256) void ln2_gate_kernel(const float* __restrict__ x2,
                                                       const float* __restrict__ g,
                                                       const float* __restrict__ b,
                                                       const float* __restrict__ wgate,
                                                       _Float16* __restrict__ h2,
                                                       float* __restrict__ maskbuf,
                                                       float* __restrict__ partials) {
    __shared__ float simp[8][3], sload[8][3];
    int wave = threadIdx.x >> 5, lane = threadIdx.x & 31;
    int row = blockIdx.x * 8 + wave;
    const float* xr = x2 + (size_t)row * 384;
    float v[12], s = 0.f;
#pragma unroll
    for (int i = 0; i < 12; i++) { v[i] = xr[i * 32 + lane]; s += v[i]; }
    s = red_sum32(s);
    float mu = s * (1.f / 384.f), qv = 0.f;
#pragma unroll
    for (int i = 0; i < 12; i++) { float d0 = v[i] - mu; qv += d0 * d0; }
    qv = red_sum32(qv);
    float rstd = rsqrtf(qv * (1.f / 384.f) + 1e-5f);
    _Float16* hr = h2 + (size_t)row * 384;
    float lg0 = 0.f, lg1 = 0.f, lg2 = 0.f;
#pragma unroll
    for (int i = 0; i < 12; i++) {
        int c = i * 32 + lane;
        float hv = (v[i] - mu) * rstd * g[c] + b[c];
        hr[c] = (_Float16)hv;
        lg0 += hv * wgate[c * 3 + 0];
        lg1 += hv * wgate[c * 3 + 1];
        lg2 += hv * wgate[c * 3 + 2];
    }
    lg0 = red_sum32(lg0); lg1 = red_sum32(lg1); lg2 = red_sum32(lg2);
    if (lane == 0) {
        float m = fmaxf(lg0, fmaxf(lg1, lg2));
        float p[3] = { __expf(lg0 - m), __expf(lg1 - m), __expf(lg2 - m) };
        float ps = p[0] + p[1] + p[2];
        p[0] /= ps; p[1] /= ps; p[2] /= ps;
        int i0 = 0;
        if (p[1] > p[i0]) i0 = 1;
        if (p[2] > p[i0]) i0 = 2;
        int i1 = -1; float best = -1.f;
#pragma unroll
        for (int e = 0; e < 3; e++)
            if (e != i0 && p[e] > best) { best = p[e]; i1 = e; }
        float gsum = p[i0] + p[i1];
        float mk[3] = {0.f, 0.f, 0.f};
        mk[i0] = p[i0] / gsum; mk[i1] = p[i1] / gsum;
#pragma unroll
        for (int e = 0; e < 3; e++) {
            maskbuf[(size_t)row * 3 + e] = mk[e];
            simp[wave][e]  = p[e];
            sload[wave][e] = (e == i0) ? 1.f : 0.f;
        }
    }
    __syncthreads();
    if (threadIdx.x == 0) {
#pragma unroll
        for (int e = 0; e < 3; e++) {
            float si = 0.f, sl = 0.f;
            for (int w = 0; w < 8; w++) { si += simp[w][e]; sl += sload[w][e]; }
            partials[(size_t)blockIdx.x * 6 + e]     = si;
            partials[(size_t)blockIdx.x * 6 + 3 + e] = sl;
        }
    }
}

__global__ __launch_bounds__(256) void moe_kernel(const _Float16* __restrict__ h2,
                                                  const _Float16* __restrict__ w1T,
                                                  const float* __restrict__ b1,
                                                  const _Float16* __restrict__ w2T,
                                                  const float* __restrict__ b2,
                                                  const float* __restrict__ maskbuf,
                                                  const float* __restrict__ x2,
                                                  float* __restrict__ out) {
    __shared__ _Float16 Hs[8][3][256];
    int wave = threadIdx.x >> 5, lane = threadIdx.x & 31;
    int gidx = blockIdx.x * 8 + wave;
    int row0 = gidx * 16;
    int rowoff = (lane >= 16) ? 8 : 0;
    const _Float16* A = h2 + (size_t)row0 * 384;
    for (int e = 0; e < 3; e++) {
        v8f acc = {};
        const _Float16* W = w1T + (size_t)e * 16 * 384;   // [16][384] K-contiguous
        for (int k0 = 0; k0 < 384; k0 += 32)
            acc = WMMA(load_a(A, 384, k0, lane), load_bk(W, 384, 0, k0, lane), acc);
        int n = lane & 15;
#pragma unroll
        for (int r = 0; r < 8; r++) {
            float hv = acc[r] + b1[e * 16 + n];
            Hs[wave][e][(r + rowoff) * 16 + n] = (_Float16)fmaxf(hv, 0.f);
        }
    }
    float wgt[8][3];
#pragma unroll
    for (int r = 0; r < 8; r++)
#pragma unroll
        for (int e = 0; e < 3; e++)
            wgt[r][e] = maskbuf[(size_t)(row0 + r + rowoff) * 3 + e];
    v16h ha[3];
#pragma unroll
    for (int e = 0; e < 3; e++) ha[e] = load_hid_a(Hs[wave][e], lane);
    for (int nt = 0; nt < 24; nt++) {
        int n = nt * 16 + (lane & 15);
        float yacc[8] = {0.f, 0.f, 0.f, 0.f, 0.f, 0.f, 0.f, 0.f};
#pragma unroll
        for (int e = 0; e < 3; e++) {
            v8f t = {};
            t = WMMA(ha[e], load_b_w2t(w2T + (size_t)e * 384 * 16, nt * 16, lane), t);
            float be = b2[e * 384 + n];
#pragma unroll
            for (int r = 0; r < 8; r++) yacc[r] += wgt[r][e] * (t[r] + be);
        }
#pragma unroll
        for (int r = 0; r < 8; r++) {
            size_t idx = (size_t)(row0 + r + rowoff) * 384 + n;
            out[idx] = x2[idx] + yacc[r];
        }
    }
}

__global__ void aux_kernel(const float* __restrict__ partials, float* __restrict__ out,
                           int nblk, int out_off) {
    if (threadIdx.x == 0 && blockIdx.x == 0) {
        float imp[3] = {0.f, 0.f, 0.f}, ld[3] = {0.f, 0.f, 0.f};
        for (int i = 0; i < nblk; i++) {
            for (int e = 0; e < 3; e++) {
                imp[e] += partials[(size_t)i * 6 + e];
                ld[e]  += partials[(size_t)i * 6 + 3 + e];
            }
        }
        float inv = 1.f / 32768.f, s = 0.f;
        for (int e = 0; e < 3; e++) s += (imp[e] * inv) * (ld[e] * inv);
        out[out_off] = 3.f * s * 0.01f;
    }
}

// ---------------- host launcher ----------------
extern "C" void kernel_launch(void* const* d_in, const int* in_sizes, int n_in,
                              void* d_out, int out_size, void* d_ws, size_t ws_size,
                              hipStream_t stream) {
    const float* x      = (const float*)d_in[0];
    const float* ln1_g  = (const float*)d_in[1];
    const float* ln1_b  = (const float*)d_in[2];
    const float* wq     = (const float*)d_in[3];
    const float* wk     = (const float*)d_in[4];
    const float* wv     = (const float*)d_in[5];
    const float* wproj  = (const float*)d_in[6];
    const float* bproj  = (const float*)d_in[7];
    const float* ln2_g  = (const float*)d_in[8];
    const float* ln2_b  = (const float*)d_in[9];
    const float* wgate  = (const float*)d_in[10];
    const float* w1     = (const float*)d_in[11];
    const float* b1     = (const float*)d_in[12];
    const float* w2     = (const float*)d_in[13];
    const float* b2     = (const float*)d_in[14];
    float* out = (float*)d_out;

    const size_t BT  = 128 * 256;        // 32768 rows
    const size_t C   = 384;
    const size_t WHD = 6 * 384 * 64;     // 147456

    char* ws = (char*)d_ws;
    size_t off = 0;
    auto alloc = [&](size_t bytes) { size_t o = off; off += (bytes + 255) & ~(size_t)255; return o; };

    _Float16* wqT   = (_Float16*)(ws + alloc(WHD * 2));       // [H][64][384]
    _Float16* wkT   = (_Float16*)(ws + alloc(WHD * 2));
    _Float16* wvT   = (_Float16*)(ws + alloc(WHD * 2));
    _Float16* wpT   = (_Float16*)(ws + alloc(C * C * 2));     // [384][384] N-major
    _Float16* w1T   = (_Float16*)(ws + alloc(3 * 16 * 384 * 2)); // [E][16][384]
    _Float16* w2T   = (_Float16*)(ws + alloc(3 * 384 * 16 * 2)); // [E][384][16]
    _Float16* h16   = (_Float16*)(ws + alloc(BT * C * 2));    // reused as attn output
    _Float16* q16   = (_Float16*)(ws + alloc(BT * C * 2));    // [B,H,T,D]; reused as h2
    _Float16* k16   = (_Float16*)(ws + alloc(BT * C * 2));
    _Float16* v16   = (_Float16*)(ws + alloc(BT * C * 2));
    float*    x2    = (float*)   (ws + alloc(BT * C * 4));
    float*    maskbf= (float*)   (ws + alloc(BT * 3 * 4));
    float*    parts = (float*)   (ws + alloc(4096 * 6 * 4));
    _Float16* attn16 = h16;   // reuse after QKV consumed h16
    _Float16* h2_16  = q16;   // reuse after attention consumed q16

    // 1. weight conversion + transpose (K becomes contiguous for B fragments)
    cvtT_kernel<<<(int)((WHD + 255) / 256), 256, 0, stream>>>(wq, wqT, 384, 64, (int)WHD);
    cvtT_kernel<<<(int)((WHD + 255) / 256), 256, 0, stream>>>(wk, wkT, 384, 64, (int)WHD);
    cvtT_kernel<<<(int)((WHD + 255) / 256), 256, 0, stream>>>(wv, wvT, 384, 64, (int)WHD);
    cvtT_kernel<<<(int)((C * C + 255) / 256), 256, 0, stream>>>(wproj, wpT, 384, 384, (int)(C * C));
    cvtT_kernel<<<(18432 + 255) / 256, 256, 0, stream>>>(w1, w1T, 384, 16, 18432);
    cvtT_kernel<<<(18432 + 255) / 256, 256, 0, stream>>>(w2, w2T, 16, 384, 18432);

    // 2. LN1
    ln1_kernel<<<4096, 256, 0, stream>>>(x, ln1_g, ln1_b, h16);
    // 3. fused QKV: 128*6*16*4 = 49152 tiles, 8 waves/block
    qkv_kernel<<<6144, 256, 0, stream>>>(h16, wqT, wkT, wvT, q16, k16, v16);
    // 4. attention: one WG per (b, head)
    attn_kernel<<<768, 256, 0, stream>>>(q16, k16, v16, attn16);
    // 5. proj + residual: 2048*24 tiles
    proj_kernel<<<6144, 256, 0, stream>>>(attn16, wpT, bproj, x, x2);
    // 6. LN2 + gating (deterministic per-block partials)
    ln2_gate_kernel<<<4096, 256, 0, stream>>>(x2, ln2_g, ln2_b, wgate, h2_16, maskbf, parts);
    // 7. MoE + residual -> out
    moe_kernel<<<256, 256, 0, stream>>>(h2_16, w1T, b1, w2T, b2, maskbf, x2, out);
    // 8. aux loss scalar
    aux_kernel<<<1, 32, 0, stream>>>(parts, out, 4096, (int)(BT * C));

    (void)in_sizes; (void)n_in; (void)out_size; (void)ws_size;
}
